// AngularPhysicsGate_28389733826621
// MI455X (gfx1250) — matrix operand order repaired and checked by reading, not compiled, not run on hardware
//
#include <hip/hip_runtime.h>
#include <hip/hip_bf16.h>
#include <math.h>

typedef _Float16 f16;
typedef __attribute__((ext_vector_type(16))) _Float16 v16h;
typedef __attribute__((ext_vector_type(8)))  _Float16 v8h;
typedef __attribute__((ext_vector_type(4)))  _Float16 v4h;
typedef __attribute__((ext_vector_type(8)))  float    v8f;

constexpr int B = 2, L = 2048, D = 2048, H = 8;
constexpr float EPS = 1e-6f;

// ---------------- Kernel 0: zero the double accumulators ----------------
__global__ void k_init(double* __restrict__ stats) {
    int t = threadIdx.x;
    if (t < 2 * B) stats[t] = 0.0;
}

// ---------------- Kernel 1: unit-normalize rows, emit f16 ----------------
// grid: B*L blocks of 256 threads; each thread handles 8 floats of a d=2048 row.
__global__ void k_normalize(const float* __restrict__ x, f16* __restrict__ uh) {
    int row = blockIdx.x;
    int t   = threadIdx.x;
    const float4* xr = (const float4*)(x + (size_t)row * D);
    float4 v0 = xr[t];
    float4 v1 = xr[t + 256];
    float ss = v0.x*v0.x + v0.y*v0.y + v0.z*v0.z + v0.w*v0.w
             + v1.x*v1.x + v1.y*v1.y + v1.z*v1.z + v1.w*v1.w;
    __shared__ float red[256];
    red[t] = ss; __syncthreads();
    for (int s = 128; s > 0; s >>= 1) { if (t < s) red[t] += red[t + s]; __syncthreads(); }
    float inv = 1.0f / fmaxf(sqrtf(red[0]), EPS);
    v4h* ur = (v4h*)(uh + (size_t)row * D);
    v4h h0 = { (f16)(v0.x*inv), (f16)(v0.y*inv), (f16)(v0.z*inv), (f16)(v0.w*inv) };
    v4h h1 = { (f16)(v1.x*inv), (f16)(v1.y*inv), (f16)(v1.z*inv), (f16)(v1.w*inv) };
    ur[t]       = h0;
    ur[t + 256] = h1;
}

// ---------------- Kernel 2: Gram matrix via WMMA + global stats ----------------
__device__ inline void store_tile(float* __restrict__ Kb, v8f c, int r0, int c0, int m,
                                  double& lsum, double& lsq) {
#pragma unroll
    for (int p = 0; p < 8; p++) {
        float v = c[p];
        Kb[(size_t)(r0 + p) * L + c0 + m] = v;
        lsum += v;
        lsq  += (double)v * (double)v;
    }
}

// Block tile 128(M) x 64(N); 8 waves in 4x2; each wave: 32x32 via 2x2 WMMA accumulators.
__global__ void k_gram(const f16* __restrict__ uh, float* __restrict__ Kraw,
                       double* __restrict__ stats) {
    int b   = blockIdx.z;
    int i0  = blockIdx.y * 128;
    int j0  = blockIdx.x * 64;
    int tid = threadIdx.x;
    int lane = tid & 31, w = tid >> 5;
    int wr = w & 3, wc = w >> 2;
    int wi = i0 + wr * 32, wj = j0 + wc * 32;
    int m = lane & 15, sel = lane >> 4;

    const f16* U   = uh + (size_t)b * L * D;
    const f16* rA0 = U + (size_t)(wi + m) * D;        // A rows wi..wi+15
    const f16* rA1 = rA0 + (size_t)16 * D;            // A rows wi+16..wi+31
    const f16* rB0 = U + (size_t)(wj + m) * D;        // B cols wj..wj+15 (= U rows)
    const f16* rB1 = rB0 + (size_t)16 * D;            // B cols wj+16..wj+31

    v8f c00 = {}, c01 = {}, c10 = {}, c11 = {};

    for (int k0 = 0; k0 < D; k0 += 32) {
        // A 16x32 f16 layout: lanes 0-15 hold K {0..15}; lanes 16-31 hold K {16..31}
        v8h a0lo = *(const v8h*)(rA0 + k0 + sel * 16);
        v8h a0hi = *(const v8h*)(rA0 + k0 + sel * 16 + 8);
        v8h a1lo = *(const v8h*)(rA1 + k0 + sel * 16);
        v8h a1hi = *(const v8h*)(rA1 + k0 + sel * 16 + 8);
        v16h A0, A1;
#pragma unroll
        for (int q = 0; q < 8; q++) {
            A0[q] = a0lo[q]; A0[q + 8] = a0hi[q];
            A1[q] = a1lo[q]; A1[q + 8] = a1hi[q];
        }
        // B 32x16 f16 layout: lane n holds column n, K = sel*16 .. sel*16+15 contiguous
        v16h B0 = *(const v16h*)(rB0 + k0 + sel * 16);
        v16h B1 = *(const v16h*)(rB1 + k0 + sel * 16);

        c00 = __builtin_amdgcn_wmma_f32_16x16x32_f16(false, A0, false, B0, (short)0, c00, false, false);
        c01 = __builtin_amdgcn_wmma_f32_16x16x32_f16(false, A0, false, B1, (short)0, c01, false, false);
        c10 = __builtin_amdgcn_wmma_f32_16x16x32_f16(false, A1, false, B0, (short)0, c10, false, false);
        c11 = __builtin_amdgcn_wmma_f32_16x16x32_f16(false, A1, false, B1, (short)0, c11, false, false);
    }

    float* Kb = Kraw + (size_t)b * L * L;
    double lsum = 0.0, lsq = 0.0;
    // C/D layout: lanes 0-15 -> rows M=vgpr, lanes 16-31 -> rows M=vgpr+8; col = lane%16
    store_tile(Kb, c00, wi      + sel * 8, wj,      m, lsum, lsq);
    store_tile(Kb, c01, wi      + sel * 8, wj + 16, m, lsum, lsq);
    store_tile(Kb, c10, wi + 16 + sel * 8, wj,      m, lsum, lsq);
    store_tile(Kb, c11, wi + 16 + sel * 8, wj + 16, m, lsum, lsq);

    __shared__ double rs[256], rq[256];
    rs[tid] = lsum; rq[tid] = lsq; __syncthreads();
    for (int s = 128; s > 0; s >>= 1) {
        if (tid < s) { rs[tid] += rs[tid + s]; rq[tid] += rq[tid + s]; }
        __syncthreads();
    }
    if (tid == 0) {
        atomicAdd(&stats[b * 2 + 0], rs[0]);
        atomicAdd(&stats[b * 2 + 1], rq[0]);
    }
}

// ---------------- Kernel 3: mean + unbiased std -> (mean, invstd) ----------------
__global__ void k_stats(const double* __restrict__ stats, float* __restrict__ mstd) {
    int b = threadIdx.x;
    if (b < B) {
        double N    = (double)L * (double)L;
        double mean = stats[b * 2 + 0] / N;
        double var  = (stats[b * 2 + 1] - N * mean * mean) / (N - 1.0);  // ddof=1
        double sd   = sqrt(var > 0.0 ? var : 0.0);
        mstd[b * 2 + 0] = (float)mean;
        mstd[b * 2 + 1] = 1.0f / fmaxf((float)sd, EPS);
    }
}

// ---------------- Kernel 4: fused attn mean + K normalize (in-place) + gate ----------------
// grid: one block per (b, i) row; single pass over the 268 MB attn tensor.
__global__ void k_epilogue(const float* __restrict__ attn, const float* __restrict__ Wp,
                           const float* __restrict__ bp, const float* __restrict__ mstd,
                           float* __restrict__ out) {
    int row = blockIdx.x;        // b*L + i
    int b = row / L, i = row % L;
    int t = threadIdx.x;

    float* gate = out;
    float* cosA = out + (size_t)B * L;
    float* Kn   = out + (size_t)2 * B * L;

    float mean = mstd[b * 2 + 0], invstd = mstd[b * 2 + 1];
    float4* Krow = (float4*)(Kn + (size_t)row * L);
    float acc = 0.0f;

    for (int c = 0; c < L / 4; c += 256) {
        int idx = c + t;
        float4 k4 = Krow[idx];
        float4 a4 = {0.f, 0.f, 0.f, 0.f};
#pragma unroll
        for (int h = 0; h < H; h++) {
            const float4* ar = (const float4*)(attn + (((size_t)(b * H + h) * L + i) * L));
            __builtin_prefetch(ar + idx + 256, 0, 0);
            float4 v = ar[idx];
            a4.x += v.x; a4.y += v.y; a4.z += v.z; a4.w += v.w;
        }
        float4 kn;
        kn.x = (k4.x - mean) * invstd;
        kn.y = (k4.y - mean) * invstd;
        kn.z = (k4.z - mean) * invstd;
        kn.w = (k4.w - mean) * invstd;
        Krow[idx] = kn;   // in-place: raw K -> K_norm
        acc += a4.x * kn.x + a4.y * kn.y + a4.z * kn.z + a4.w * kn.w;
    }
    acc *= (1.0f / (float)H);    // head mean

    __shared__ float red[256];
    red[t] = acc; __syncthreads();
    for (int s = 128; s > 0; s >>= 1) { if (t < s) red[t] += red[t + s]; __syncthreads(); }
    if (t == 0) {
        float ca = red[0];
        float z  = Wp[0] * ca + bp[0];
        gate[row] = 1.0f / (1.0f + __expf(-z));
        cosA[row] = ca;
    }
}

extern "C" void kernel_launch(void* const* d_in, const int* in_sizes, int n_in,
                              void* d_out, int out_size, void* d_ws, size_t ws_size,
                              hipStream_t stream) {
    const float* x    = (const float*)d_in[0];  // [B,L,D]
    const float* attn = (const float*)d_in[1];  // [B,H,L,L]
    const float* Wp   = (const float*)d_in[2];  // [1]
    const float* bp   = (const float*)d_in[3];  // [1]
    float* out = (float*)d_out;                 // gate[B*L] | cos[B*L] | K_norm[B*L*L]

    // workspace: f16 unit vectors (16.8 MB) + 4 doubles + 4 floats
    f16*    uh    = (f16*)d_ws;
    double* stats = (double*)((char*)d_ws + (size_t)B * L * D * sizeof(f16));
    float*  mstd  = (float*)(stats + 2 * B);

    // raw K goes straight into the K_norm output slot; normalized in-place by k_epilogue
    float* Kraw = out + (size_t)2 * B * L;

    k_init<<<1, 32, 0, stream>>>(stats);
    k_normalize<<<B * L, 256, 0, stream>>>(x, uh);
    dim3 gg(L / 64, L / 128, B);
    k_gram<<<gg, 256, 0, stream>>>(uh, Kraw, stats);
    k_stats<<<1, 32, 0, stream>>>(stats, mstd);
    k_epilogue<<<B * L, 256, 0, stream>>>(attn, Wp, bp, mstd, out);
}
